// CrossAttentionLayer_45732811768487
// MI455X (gfx1250) — compile-verified, matrix operand order checked by reading, not run on hardware
//
#include <hip/hip_runtime.h>
#include <hip/hip_bf16.h>

// ---------------------------------------------------------------------------
// CDNA5 (gfx1250) ragged cross-attention layer.
// Wave32, f16 WMMA (V_WMMA_F32_16X16X32_F16) with fp32 accumulation.
// ---------------------------------------------------------------------------

typedef __attribute__((ext_vector_type(16))) _Float16 v16h;
typedef __attribute__((ext_vector_type(8)))  _Float16 v8h;
typedef __attribute__((ext_vector_type(4)))  _Float16 v4h;
typedef __attribute__((ext_vector_type(8)))  float    v8f;

union F16x16 { v16h v; v8h h[2]; };

#define H_DIM  512
#define NHEADS 8
#define BN_EPS 1e-5f

// Fragment loader. 16-bit WMMA fragment layout (ISA 7.12.2):
//   lanes 0-15  : K = kk + {0..7}  and kk + {16..23}
//   lanes 16-31 : K = kk + {8..15} and kk + {24..31}
// `base` must already include the per-lane +8 half-select (hsel).
__device__ __forceinline__ v16h load_frag16(const _Float16* base, int kk) {
  F16x16 f;
  f.h[0] = *(const v8h*)(base + kk);
  f.h[1] = *(const v8h*)(base + kk + 16);
  return f.v;
}

// ---------------------------------------------------------------------------
// fp32 -> f16 conversion (4 elements / thread)
// ---------------------------------------------------------------------------
__global__ void cvt_f16_kernel(const float* __restrict__ in,
                               _Float16* __restrict__ out, int n) {
  int i = (blockIdx.x * blockDim.x + threadIdx.x) * 4;
  if (i + 3 < n) {
    float4 f = *(const float4*)(in + i);
    v4h h;
    h.x = (_Float16)f.x; h.y = (_Float16)f.y;
    h.z = (_Float16)f.z; h.w = (_Float16)f.w;
    *(v4h*)(out + i) = h;
  } else {
    for (int j = i; j < n; ++j) out[j] = (_Float16)in[j];
  }
}

// dst_h fp32 [TD x 512] -> X f16 [TD x 1024] columns 0..511 (concat left half)
__global__ void pack_dst_kernel(const float* __restrict__ in,
                                _Float16* __restrict__ X, int total) {
  int i = blockIdx.x * blockDim.x + threadIdx.x;
  if (i < total) {
    int r = i >> 9, c = i & 511;
    X[(size_t)r * 1024 + c] = (_Float16)in[i];
  }
}

__global__ void init_stats_kernel(float* __restrict__ s) {
  int i = blockIdx.x * blockDim.x + threadIdx.x;
  if (i < 2048) s[i] = 0.f;
}

// ---------------------------------------------------------------------------
// Generic WMMA GEMM:  C[M x N] = A[M x K] @ W[N x K]^T + bias
// A, W are f16 row-major (K contiguous).  One wave -> 32x64 output tile:
// 2 A fragments x 4 B fragments -> 8 WMMAs per K=32 step, 6 fragment loads
// (1.5 b128 loads per WMMA; two independent WMMA chains per B fragment).
// Output modes:
//   0: f16, out[m*opitch + ocoloff + n]
//   1: f16 head-major rows     Qh/Kh: out[(h*M + m)*64 + d]   (c = d*8+h)
//   2: f16 head-major s-major  Vt:    out[(h*64 + d)*M + m]
//   3: f32 row-major           out[m*N + n]
// ---------------------------------------------------------------------------
__global__ __launch_bounds__(256)
void gemm_f16_kernel(const _Float16* __restrict__ A, int lda,
                     const _Float16* __restrict__ W,
                     const float* __restrict__ bias,
                     void* __restrict__ out,
                     int M, int N, int K,
                     int mode, int opitch, int ocoloff) {
  const int lane = threadIdx.x & 31;
  const int wid  = threadIdx.x >> 5;
  const int tm   = blockIdx.x * 8 + wid;        // 32-row tile index
  if (tm * 32 >= M) return;
  const int m0   = tm * 32;
  const bool has2 = (m0 + 16) < M;              // second 16-row sub-tile valid?
  const int n0   = blockIdx.y * 64;
  const int lo   = lane & 15;
  const int hsel = (lane & 16) ? 8 : 0;

  const _Float16* arow0 = A + (size_t)(m0 + lo) * lda + hsel;
  const _Float16* arow1 = A + (size_t)(m0 + (has2 ? 16 : 0) + lo) * lda + hsel;
  const _Float16* w0    = W + (size_t)(n0 + lo) * K + hsel;
  const size_t wstep    = (size_t)16 * K;

  v8f acc[8] = {};   // [0..3] = rows m0..m0+15, [4..7] = rows m0+16..m0+31
  for (int kk = 0; kk < K; kk += 32) {
    if (kk + 64 < K) __builtin_prefetch(arow0 + kk + 64, 0, 1);
    v16h a0 = load_frag16(arow0, kk);
    v16h a1 = load_frag16(arow1, kk);
    v16h b0 = load_frag16(w0,             kk);
    v16h b1 = load_frag16(w0 + wstep,     kk);
    v16h b2 = load_frag16(w0 + 2 * wstep, kk);
    v16h b3 = load_frag16(w0 + 3 * wstep, kk);
    acc[0] = __builtin_amdgcn_wmma_f32_16x16x32_f16(false, a0, false, b0, (short)0, acc[0], false, false);
    acc[4] = __builtin_amdgcn_wmma_f32_16x16x32_f16(false, a1, false, b0, (short)0, acc[4], false, false);
    acc[1] = __builtin_amdgcn_wmma_f32_16x16x32_f16(false, a0, false, b1, (short)0, acc[1], false, false);
    acc[5] = __builtin_amdgcn_wmma_f32_16x16x32_f16(false, a1, false, b1, (short)0, acc[5], false, false);
    acc[2] = __builtin_amdgcn_wmma_f32_16x16x32_f16(false, a0, false, b2, (short)0, acc[2], false, false);
    acc[6] = __builtin_amdgcn_wmma_f32_16x16x32_f16(false, a1, false, b2, (short)0, acc[6], false, false);
    acc[3] = __builtin_amdgcn_wmma_f32_16x16x32_f16(false, a0, false, b3, (short)0, acc[3], false, false);
    acc[7] = __builtin_amdgcn_wmma_f32_16x16x32_f16(false, a1, false, b3, (short)0, acc[7], false, false);
  }

  // C/D layout: VGPR r, lanes 0-15 -> M=base+r, N=lo; lanes 16-31 -> M=base+r+8
  const int mofs = (lane & 16) ? 8 : 0;
#pragma unroll
  for (int half = 0; half < 2; ++half) {
    if (half == 1 && !has2) break;
    const int mbase = m0 + 16 * half + mofs;
#pragma unroll
    for (int t = 0; t < 4; ++t) {
      const int n = n0 + 16 * t + lo;
      const float bv = bias[n];
      const v8f av = acc[half * 4 + t];
#pragma unroll
      for (int r = 0; r < 8; ++r) {
        const int m = mbase + r;
        float v = av[r] + bv;
        if (mode == 0) {
          ((_Float16*)out)[(size_t)m * opitch + ocoloff + n] = (_Float16)v;
        } else if (mode == 1) {
          int hh = n & 7, d = n >> 3;
          ((_Float16*)out)[((size_t)hh * M + m) * 64 + d] = (_Float16)v;
        } else if (mode == 2) {
          int hh = n & 7, d = n >> 3;
          ((_Float16*)out)[(size_t)(hh * 64 + d) * M + m] = (_Float16)v;
        } else {
          ((float*)out)[(size_t)m * N + n] = v;
        }
      }
    }
  }
}

// ---------------------------------------------------------------------------
// Flash-style ragged attention.  Grid: (TD/16, NHEADS), 32 threads (1 wave).
// Each wave: 16 dst rows x one head.  Loops src in 32-col chunks:
//   S = Q Kh^T / 8  (2 C-frags, 4 WMMAs), online softmax, P -> LDS bounce
//   (C-layout -> A-layout), O += P @ V (4 WMMAs against s-major Vt).
// Writes Omsg f16 [TD x 512], channel c = d*8 + h.
// ---------------------------------------------------------------------------
__global__ __launch_bounds__(32)
void attn_kernel(const _Float16* __restrict__ Qh, const _Float16* __restrict__ Kh,
                 const _Float16* __restrict__ Vt, _Float16* __restrict__ Omsg,
                 const int* __restrict__ dlens, const int* __restrict__ slens,
                 int TD, int TS) {
  __shared__ _Float16 ldsP[16 * 32];

  const int lane = threadIdx.x;
  const int h    = blockIdx.y;
  const int lo   = lane & 15;
  const int hi   = (lane >> 4) & 1;
  const int hsel = hi * 8;

  // locate (batch, tile) for this block in the ragged dst layout
  int t = blockIdx.x, b = 0, offd = 0, offs = 0;
  for (;;) {
    int ld = dlens[b];
    int nt = (ld + 15) >> 4;
    if (t < nt) break;
    t -= nt; offd += ld; offs += slens[b];
    if (++b >= 16) return;
  }
  const int Ld = dlens[b], Ls = slens[b];
  const int n0 = t * 16;

  // Q A-fragments for this 16-row tile (d = 0..63 -> two K=32 fragments)
  int qrow = n0 + lo; if (qrow >= Ld) qrow = Ld - 1;
  const _Float16* qbase = Qh + ((size_t)h * TD + (offd + qrow)) * 64 + hsel;
  v16h aq0 = load_frag16(qbase, 0);
  v16h aq1 = load_frag16(qbase, 32);

  float m_i[8], l_i[8];
#pragma unroll
  for (int r = 0; r < 8; ++r) { m_i[r] = -1e30f; l_i[r] = 0.f; }
  v8f o0 = {}, o1 = {}, o2 = {}, o3 = {};

  for (int s0 = 0; s0 < Ls; s0 += 32) {
    // ---- S tiles: columns [s0, s0+16) and [s0+16, s0+32) ----
    v8f sc0 = {}, sc1 = {};
    {
      int sc = s0 + lo; if (sc >= Ls) sc = Ls - 1;
      const _Float16* kb = Kh + ((size_t)h * TS + offs + sc) * 64 + hsel;
      v16h b0 = load_frag16(kb, 0), b1 = load_frag16(kb, 32);
      sc0 = __builtin_amdgcn_wmma_f32_16x16x32_f16(false, aq0, false, b0, (short)0, sc0, false, false);
      sc0 = __builtin_amdgcn_wmma_f32_16x16x32_f16(false, aq1, false, b1, (short)0, sc0, false, false);
    }
    {
      int sc = s0 + 16 + lo; if (sc >= Ls) sc = Ls - 1;
      const _Float16* kb = Kh + ((size_t)h * TS + offs + sc) * 64 + hsel;
      v16h b0 = load_frag16(kb, 0), b1 = load_frag16(kb, 32);
      sc1 = __builtin_amdgcn_wmma_f32_16x16x32_f16(false, aq0, false, b0, (short)0, sc1, false, false);
      sc1 = __builtin_amdgcn_wmma_f32_16x16x32_f16(false, aq1, false, b1, (short)0, sc1, false, false);
    }
    const bool inv0 = (s0 + lo) >= Ls;
    const bool inv1 = (s0 + 16 + lo) >= Ls;

    // ---- online softmax (row stats live in 16-lane half-groups) ----
    float scale[8];
#pragma unroll
    for (int r = 0; r < 8; ++r) {
      float v0 = inv0 ? -1e30f : sc0[r] * 0.125f;
      float v1 = inv1 ? -1e30f : sc1[r] * 0.125f;
      float mx = fmaxf(v0, v1);
      mx = fmaxf(mx, __shfl_xor(mx, 1, 32));
      mx = fmaxf(mx, __shfl_xor(mx, 2, 32));
      mx = fmaxf(mx, __shfl_xor(mx, 4, 32));
      mx = fmaxf(mx, __shfl_xor(mx, 8, 32));
      float mn = fmaxf(m_i[r], mx);
      float sc = __expf(m_i[r] - mn);
      float p0 = __expf(v0 - mn);
      float p1 = __expf(v1 - mn);
      float rs = p0 + p1;
      rs += __shfl_xor(rs, 1, 32);
      rs += __shfl_xor(rs, 2, 32);
      rs += __shfl_xor(rs, 4, 32);
      rs += __shfl_xor(rs, 8, 32);
      l_i[r] = l_i[r] * sc + rs;
      m_i[r] = mn;
      scale[r] = sc;
      const int m = r + 8 * hi;                 // row within tile (C layout)
      ldsP[m * 32 + lo]      = (_Float16)p0;    // chunk col 16*0 + lo
      ldsP[m * 32 + 16 + lo] = (_Float16)p1;    // chunk col 16*1 + lo
    }
    __syncthreads();

    // rescale running O accumulators
#pragma unroll
    for (int r = 0; r < 8; ++r) {
      o0[r] *= scale[r]; o1[r] *= scale[r];
      o2[r] *= scale[r]; o3[r] *= scale[r];
    }

    // P back as A-fragment (K = 32 src cols)
    v16h ap = load_frag16(ldsP + lo * 32 + hsel, 0);

    // V B-fragments from s-major Vt: two contiguous 8-half runs per lane
    int s_a = s0 + hsel;      if (s_a >= Ls) s_a = Ls - 8;   // 8-aligned chunks
    int s_b = s0 + hsel + 16; if (s_b >= Ls) s_b = Ls - 8;
    {
      const _Float16* vb = Vt + ((size_t)(h * 64 + 0 + lo)) * TS + offs;
      F16x16 f; f.h[0] = *(const v8h*)(vb + s_a); f.h[1] = *(const v8h*)(vb + s_b);
      o0 = __builtin_amdgcn_wmma_f32_16x16x32_f16(false, ap, false, f.v, (short)0, o0, false, false);
    }
    {
      const _Float16* vb = Vt + ((size_t)(h * 64 + 16 + lo)) * TS + offs;
      F16x16 f; f.h[0] = *(const v8h*)(vb + s_a); f.h[1] = *(const v8h*)(vb + s_b);
      o1 = __builtin_amdgcn_wmma_f32_16x16x32_f16(false, ap, false, f.v, (short)0, o1, false, false);
    }
    {
      const _Float16* vb = Vt + ((size_t)(h * 64 + 32 + lo)) * TS + offs;
      F16x16 f; f.h[0] = *(const v8h*)(vb + s_a); f.h[1] = *(const v8h*)(vb + s_b);
      o2 = __builtin_amdgcn_wmma_f32_16x16x32_f16(false, ap, false, f.v, (short)0, o2, false, false);
    }
    {
      const _Float16* vb = Vt + ((size_t)(h * 64 + 48 + lo)) * TS + offs;
      F16x16 f; f.h[0] = *(const v8h*)(vb + s_a); f.h[1] = *(const v8h*)(vb + s_b);
      o3 = __builtin_amdgcn_wmma_f32_16x16x32_f16(false, ap, false, f.v, (short)0, o3, false, false);
    }
    __syncthreads();
  }

  // normalize + scatter to channel-major Omsg (c = d*8 + h)
#pragma unroll
  for (int r = 0; r < 8; ++r) {
    const int mloc = n0 + r + 8 * hi;
    if (mloc >= Ld) continue;
    const float inv = 1.f / l_i[r];
    const size_t row = (size_t)(offd + mloc) * 512;
    Omsg[row + ((0  + lo) << 3) + h] = (_Float16)(o0[r] * inv);
    Omsg[row + ((16 + lo) << 3) + h] = (_Float16)(o1[r] * inv);
    Omsg[row + ((32 + lo) << 3) + h] = (_Float16)(o2[r] * inv);
    Omsg[row + ((48 + lo) << 3) + h] = (_Float16)(o3[r] * inv);
  }
}

// ---------------------------------------------------------------------------
// BatchNorm (training-mode batch stats) helpers
// ---------------------------------------------------------------------------
__global__ __launch_bounds__(256)
void bn_stats_kernel(const float* __restrict__ x, int M,
                     float* __restrict__ sum, float* __restrict__ sq) {
  const int c0 = threadIdx.x;           // channels c0 and c0+256
  float s0 = 0.f, s1 = 0.f, q0 = 0.f, q1 = 0.f;
  for (int r = blockIdx.x; r < M; r += gridDim.x) {
    const float* row = x + (size_t)r * 512;
    float a = row[c0], b = row[c0 + 256];
    s0 += a; q0 += a * a; s1 += b; q1 += b * b;
  }
  atomicAdd(&sum[c0], s0);        atomicAdd(&sq[c0], q0);
  atomicAdd(&sum[c0 + 256], s1);  atomicAdd(&sq[c0 + 256], q1);
}

__global__ void bn_relu_f16_kernel(const float* __restrict__ x,
                                   const float* __restrict__ sum,
                                   const float* __restrict__ sq,
                                   const float* __restrict__ g,
                                   const float* __restrict__ be,
                                   int M, _Float16* __restrict__ Y) {
  int i = blockIdx.x * blockDim.x + threadIdx.x;
  if (i >= M * 512) return;
  int c = i & 511;
  float mean = sum[c] / (float)M;
  float var  = sq[c] / (float)M - mean * mean;
  float v = (x[i] - mean) * rsqrtf(var + BN_EPS) * g[c] + be[c];
  Y[i] = (_Float16)fmaxf(v, 0.f);
}

__global__ void bn_residual_kernel(const float* __restrict__ z,
                                   const float* __restrict__ sum,
                                   const float* __restrict__ sq,
                                   const float* __restrict__ g,
                                   const float* __restrict__ be,
                                   const float* __restrict__ resid,
                                   int M, float* __restrict__ out) {
  int i = blockIdx.x * blockDim.x + threadIdx.x;
  if (i >= M * 512) return;
  int c = i & 511;
  float mean = sum[c] / (float)M;
  float var  = sq[c] / (float)M - mean * mean;
  out[i] = resid[i] + (z[i] - mean) * rsqrtf(var + BN_EPS) * g[c] + be[c];
}

// ---------------------------------------------------------------------------
// Host launcher
// ---------------------------------------------------------------------------
extern "C" void kernel_launch(void* const* d_in, const int* in_sizes, int n_in,
                              void* d_out, int out_size, void* d_ws, size_t ws_size,
                              hipStream_t stream) {
  const float* src_h = (const float*)d_in[0];
  const float* dst_h = (const float*)d_in[1];
  const int*   snv   = (const int*)d_in[2];
  const int*   dnv   = (const int*)d_in[3];
  const float* Wq = (const float*)d_in[4];  const float* bq  = (const float*)d_in[5];
  const float* Wk = (const float*)d_in[6];  const float* bk  = (const float*)d_in[7];
  const float* Wv = (const float*)d_in[8];  const float* bv  = (const float*)d_in[9];
  const float* Wm = (const float*)d_in[10]; const float* bm  = (const float*)d_in[11];
  const float* W1 = (const float*)d_in[12]; const float* b1  = (const float*)d_in[13];
  const float* g1 = (const float*)d_in[14]; const float* be1 = (const float*)d_in[15];
  const float* W2 = (const float*)d_in[16]; const float* b2  = (const float*)d_in[17];
  const float* g2 = (const float*)d_in[18]; const float* be2 = (const float*)d_in[19];
  float* out = (float*)d_out;

  const int TS = in_sizes[0] / H_DIM;   // total src rows (7232)
  const int TD = in_sizes[1] / H_DIM;   // total dst rows (6016)

  // workspace layout (256B-aligned bump allocator)
  char* p = (char*)d_ws;
  auto bump = [&](size_t bytes) -> void* {
    void* r = (void*)p; p += (bytes + 255) & ~(size_t)255; return r;
  };
  _Float16* hs16 = (_Float16*)bump((size_t)TS * 512 * 2);
  _Float16* hd16 = (_Float16*)bump((size_t)TD * 512 * 2);
  _Float16* wq16 = (_Float16*)bump((size_t)512 * 512 * 2);
  _Float16* wk16 = (_Float16*)bump((size_t)512 * 512 * 2);
  _Float16* wv16 = (_Float16*)bump((size_t)512 * 512 * 2);
  _Float16* wm16 = (_Float16*)bump((size_t)512 * 512 * 2);
  _Float16* w116 = (_Float16*)bump((size_t)512 * 1024 * 2);
  _Float16* w216 = (_Float16*)bump((size_t)512 * 512 * 2);
  _Float16* Qh   = (_Float16*)bump((size_t)TD * 512 * 2);
  _Float16* Kh   = (_Float16*)bump((size_t)TS * 512 * 2);
  _Float16* Vt   = (_Float16*)bump((size_t)TS * 512 * 2);
  _Float16* Omsg = (_Float16*)bump((size_t)TD * 512 * 2);
  _Float16* X    = (_Float16*)bump((size_t)TD * 1024 * 2);
  float*    x1f  = (float*)bump((size_t)TD * 512 * 4);   // reused as z
  _Float16* Yf   = (_Float16*)bump((size_t)TD * 512 * 2);
  float*    stats = (float*)bump(2048 * 4);
  float *sum1 = stats, *sq1 = stats + 512, *sum2 = stats + 1024, *sq2 = stats + 1536;

  auto cvt = [&](const float* src, _Float16* dst, int n) {
    cvt_f16_kernel<<<(n + 1023) / 1024, 256, 0, stream>>>(src, dst, n);
  };

  init_stats_kernel<<<8, 256, 0, stream>>>(stats);

  // fp32 -> f16 staging
  cvt(src_h, hs16, TS * 512);
  cvt(dst_h, hd16, TD * 512);
  cvt(Wq, wq16, 512 * 512);
  cvt(Wk, wk16, 512 * 512);
  cvt(Wv, wv16, 512 * 512);
  cvt(Wm, wm16, 512 * 512);
  cvt(W1, w116, 512 * 1024);
  cvt(W2, w216, 512 * 512);
  pack_dst_kernel<<<(TD * 512 + 255) / 256, 256, 0, stream>>>(dst_h, X, TD * 512);

  auto gemm = [&](const _Float16* A, int lda, const _Float16* W, const float* bias,
                  void* O, int M, int N, int K, int mode, int opitch, int ocoloff) {
    const int tiles32 = (M + 31) / 32;
    dim3 grid((tiles32 + 7) / 8, N / 64);
    gemm_f16_kernel<<<grid, 256, 0, stream>>>(A, lda, W, bias, O, M, N, K,
                                              mode, opitch, ocoloff);
  };

  // projections into attention-friendly layouts
  gemm(hd16, 512, wq16, bq, Qh, TD, 512, 512, /*mode*/1, 0, 0);  // Qh[h][n][d]
  gemm(hs16, 512, wk16, bk, Kh, TS, 512, 512, /*mode*/1, 0, 0);  // Kh[h][s][d]
  gemm(hs16, 512, wv16, bv, Vt, TS, 512, 512, /*mode*/2, 0, 0);  // Vt[h][d][s]

  // ragged flash attention
  attn_kernel<<<dim3(TD / 16, NHEADS), 32, 0, stream>>>(Qh, Kh, Vt, Omsg,
                                                        dnv, snv, TD, TS);

  // merge projection straight into concat buffer X[:, 512:1024]
  gemm(Omsg, 512, wm16, bm, X, TD, 512, 512, /*mode*/0, 1024, 512);

  // MLP: x1 = X @ W1^T + b1  (K = 1024), fp32 out
  gemm(X, 1024, w116, b1, x1f, TD, 512, 1024, /*mode*/3, 0, 0);

  // BN1 + ReLU -> f16
  bn_stats_kernel<<<64, 256, 0, stream>>>(x1f, TD, sum1, sq1);
  bn_relu_f16_kernel<<<(TD * 512 + 255) / 256, 256, 0, stream>>>(
      x1f, sum1, sq1, g1, be1, TD, Yf);

  // z = Y @ W2^T + b2 (fp32 out, reuses x1f storage)
  gemm(Yf, 512, w216, b2, x1f, TD, 512, 512, /*mode*/3, 0, 0);

  // BN2 + residual
  bn_stats_kernel<<<64, 256, 0, stream>>>(x1f, TD, sum2, sq2);
  bn_residual_kernel<<<(TD * 512 + 255) / 256, 256, 0, stream>>>(
      x1f, sum2, sq2, g2, be2, dst_h, TD, out);
}
